// cross_att_34900904247517
// MI455X (gfx1250) — matrix operand order repaired and checked by reading, not compiled, not run on hardware
//
#include <hip/hip_runtime.h>
#include <cstdint>
#include <cstddef>

#define DEV __device__ __forceinline__

typedef __bf16 bf16;
typedef __attribute__((ext_vector_type(16))) __bf16 v16bf;
typedef __attribute__((ext_vector_type(8)))  __bf16 v8bf;
typedef __attribute__((ext_vector_type(8)))  float  v8f;
typedef __attribute__((ext_vector_type(4)))  float  v4f;
typedef unsigned int v4u __attribute__((ext_vector_type(4)));
typedef int          v8i __attribute__((ext_vector_type(8)));
typedef int          v4i __attribute__((ext_vector_type(4)));

static constexpr int Bb   = 8;
static constexpr int L1   = 2048;
static constexpr int L2   = 2048;
static constexpr int DIN  = 512;
static constexpr int DHID = 512;
static constexpr int DCAT = DIN + DHID;     // 1024
static constexpr int MTOT = Bb * L1;        // 16384

// Attention LDS layout (dynamic shared):
//   kvr[2]: row-major in2 chunk, 32 rows, TDM-padded stride 528 bf16 (1056B)
//   kvt   : transposed chunk [512][40]
//   pbuf  : per-wave P tiles  [4][16][40]
static constexpr int    KVR_STRIDE = 528;              // 512 + 16 pad elems
static constexpr size_t SMEM_KVR0  = 0;
static constexpr size_t SMEM_KVR1  = 32 * KVR_STRIDE * 2;            // 33792
static constexpr size_t SMEM_KVT   = 2 * 32 * KVR_STRIDE * 2;        // 67584
static constexpr int    KVT_STRIDE = 40;
static constexpr size_t SMEM_PBUF  = SMEM_KVT + 512 * KVT_STRIDE * 2; // 108544
static constexpr size_t SMEM_TOTAL = SMEM_PBUF + 4 * 16 * KVT_STRIDE * 2; // 113664

#if defined(__has_builtin)
#if __has_builtin(__builtin_amdgcn_tensor_load_to_lds) && \
    __has_builtin(__builtin_amdgcn_s_wait_tensorcnt)
#define USE_TDM 1
#endif
#endif
#ifndef USE_TDM
#define USE_TDM 0
#endif

// ---- helpers -------------------------------------------------------------

DEV bf16 f2bf(float f) {                    // round-to-nearest-even f32 -> bf16
  unsigned u = __builtin_bit_cast(unsigned, f);
  u = (u + 0x7FFFu + ((u >> 16) & 1u)) >> 16;
  unsigned short s = (unsigned short)u;
  return __builtin_bit_cast(bf16, s);
}

// A-operand (and split-B from padded LDS): two 16B runs -> 8 VGPRs
DEV v16bf load_pair(const bf16* p0, const bf16* p1) {
  v8bf lo = *reinterpret_cast<const v8bf*>(p0);
  v8bf hi = *reinterpret_cast<const v8bf*>(p1);
  v16bf r;
#pragma unroll
  for (int i = 0; i < 8; ++i) { r[i] = lo[i]; r[8 + i] = hi[i]; }
  return r;
}

// B-operand from a 32B-aligned contiguous run
DEV v16bf load_b(const bf16* p) { return *reinterpret_cast<const v16bf*>(p); }

DEV v8f wmma_bf16(v16bf a, v16bf b, v8f c) {
  return __builtin_amdgcn_wmma_f32_16x16x32_bf16(false, a, false, b, (short)0, c,
                                                 false, false);
}

#if USE_TDM
// Issue one TDM load of a 32x512 bf16 row-major tile into LDS, with the TDM
// inserting 32B of padding per 1024B row (pad_interval=256 dwords, amount=8).
// D# fields per CDNA5 ISA (group0 128b, group1 256b).
DEV void tdm_load_chunk(const bf16* gsrc, unsigned lds_off) {
  unsigned long long ga = (unsigned long long)(uintptr_t)gsrc;
  v4u g0;
  g0[0] = 1u;                                   // count=1, user mode, no gather
  g0[1] = lds_off;                              // lds_addr (bytes)
  g0[2] = (unsigned)(ga & 0xFFFFFFFFu);         // global_addr[31:0]
  g0[3] = (unsigned)((ga >> 32) & 0x01FFFFFFu)  // global_addr[56:32]
          | 0x80000000u;                        // type=2 ("image")
  v8i g1;
  g1[0] = (int)((1u << 16)                      // data_size = 2 bytes
                | (1u << 20)                    // pad_enable
                | (7u << 22)                    // pad_interval: 256 dwords
                | (7u << 25));                  // pad_amount:   8 dwords (32B)
  g1[1] = (int)(512u << 16);                    // tensor_dim0 = 512  (bits 79:48)
  g1[2] = (int)(32u << 16);                     // tensor_dim1 = 32   (bits 111:80)
  g1[3] = (int)(512u << 16);                    // tile_dim0   = 512  (bits 127:112)
  g1[4] = (int)32;                              // tile_dim1 = 32, tile_dim2 = 0
  g1[5] = (int)512;                             // tensor_dim0_stride = 512
  g1[6] = 0;                                    // stride0 hi | stride1 lo
  g1[7] = 0;
  v4i z4 = {0, 0, 0, 0};
#if defined(__clang_major__) && (__clang_major__ >= 23)
  v8i z8 = {0, 0, 0, 0, 0, 0, 0, 0};
  __builtin_amdgcn_tensor_load_to_lds(g0, g1, z4, z4, z8, 0);
#else
  __builtin_amdgcn_tensor_load_to_lds(g0, g1, z4, z4, 0);
#endif
}
#endif

// ---- kernel 0: f32 -> bf16 convert (vectorized) --------------------------

__global__ void __launch_bounds__(256) k_cvt_bf16(const float* __restrict__ src,
                                                  bf16* __restrict__ dst, int n4) {
  int i = blockIdx.x * blockDim.x + threadIdx.x;
  if (i >= n4) return;
  v4f v = reinterpret_cast<const v4f*>(src)[i];
  struct alignas(8) P4 { bf16 x[4]; };
  P4 o;
#pragma unroll
  for (int j = 0; j < 4; ++j) o.x[j] = f2bf(v[j]);
  reinterpret_cast<P4*>(dst)[i] = o;
}

// ---- kernel 1: Q = in1 @ W1^T  (2-stage software pipeline) ---------------

__global__ void __launch_bounds__(128) k_gemm_q(const bf16* __restrict__ A,   // [MTOT][512]
                                                const bf16* __restrict__ W,   // [512][512]
                                                bf16* __restrict__ Q) {       // [MTOT][512]
  int wave = threadIdx.x >> 5, lane = threadIdx.x & 31;
  int gw = blockIdx.x * 4 + wave;
  int strip = gw >> 3, nb = gw & 7;
  int m0 = strip * 16, n0 = nb * 64;
  int c = lane & 15, g = lane >> 4;

  const bf16* arow = A + (size_t)(m0 + c) * DIN;
  const bf16* brow[4];
#pragma unroll
  for (int nt = 0; nt < 4; ++nt) brow[nt] = W + (size_t)(n0 + nt * 16 + c) * DIN;

  v8f zero = {0, 0, 0, 0, 0, 0, 0, 0};
  v8f acc[4] = {zero, zero, zero, zero};

  v16bf a_cur = load_pair(arow + g * 16, arow + g * 16 + 8);
  v16bf b_cur[4];
#pragma unroll
  for (int nt = 0; nt < 4; ++nt) b_cur[nt] = load_b(brow[nt] + g * 16);

#pragma unroll
  for (int k = 0; k < DIN / 32; ++k) {
    v16bf a_nxt = a_cur;
    v16bf b_nxt[4] = {b_cur[0], b_cur[1], b_cur[2], b_cur[3]};
    if (k + 1 < DIN / 32) {                       // prefetch next K-slice
      int da = (k + 1) * 32 + g * 16, db = (k + 1) * 32 + g * 16;
      a_nxt = load_pair(arow + da, arow + da + 8);
#pragma unroll
      for (int nt = 0; nt < 4; ++nt) b_nxt[nt] = load_b(brow[nt] + db);
    }
#pragma unroll
    for (int nt = 0; nt < 4; ++nt) acc[nt] = wmma_bf16(a_cur, b_cur[nt], acc[nt]);
    a_cur = a_nxt;
#pragma unroll
    for (int nt = 0; nt < 4; ++nt) b_cur[nt] = b_nxt[nt];
  }
#pragma unroll
  for (int nt = 0; nt < 4; ++nt)
#pragma unroll
    for (int i = 0; i < 8; ++i)
      Q[(size_t)(m0 + g * 8 + i) * DHID + n0 + nt * 16 + c] = f2bf(acc[nt][i]);
}

// ---- kernel 2: fused masked cross-attention (flash, online softmax) ------
// Block = 4 waves = 2 row-strips x 2 d-halves. in2 chunks arrive via the
// Tensor Data Mover into double-buffered LDS, overlapping DMA with WMMA.

__global__ void __launch_bounds__(128) k_attn(const bf16* __restrict__ Q,
                                              const bf16* __restrict__ KV,     // in2 bf16
                                              const unsigned char* __restrict__ mask,
                                              bf16* __restrict__ fused) {
  extern __shared__ __align__(16) char smem[];
  bf16* kvr0 = (bf16*)(smem + SMEM_KVR0);
  bf16* kvr1 = (bf16*)(smem + SMEM_KVR1);
  bf16* kvt  = (bf16*)(smem + SMEM_KVT);
  bf16* pbuf = (bf16*)(smem + SMEM_PBUF);

  int b = blockIdx.y;
  int wave = threadIdx.x >> 5, lane = threadIdx.x & 31;
  int c = lane & 15, g = lane >> 4;
  int strip = blockIdx.x * 2 + (wave >> 1);
  int dhalf = wave & 1;
  int m0 = strip * 16;

  const bf16* qrow = Q + (size_t)(b * L1 + m0 + c) * DHID;
  const bf16* kvb  = KV + (size_t)b * L2 * DHID;
  const unsigned char* mrow = mask + (size_t)b * L2;
  const float NI = -__builtin_inff();
  const int NCH = L2 / 32;

  v8f zero = {0, 0, 0, 0, 0, 0, 0, 0};
  v8f o[16];
  float rm[8], rs[8];
#pragma unroll
  for (int t = 0; t < 16; ++t) o[t] = zero;
#pragma unroll
  for (int i = 0; i < 8; ++i) { rm[i] = NI; rs[i] = 0.f; }

#if USE_TDM
  if (wave == 0) tdm_load_chunk(kvb, (unsigned)(uintptr_t)kvr0);
#endif

  for (int ch = 0; ch < NCH; ++ch) {
#if USE_TDM
    bf16* kvcur = (ch & 1) ? kvr1 : kvr0;
    __syncthreads();                       // previous chunk fully consumed
    if (wave == 0) {
      if (ch + 1 < NCH) {                  // prefetch next chunk into other buffer
        tdm_load_chunk(kvb + (size_t)(ch + 1) * 32 * DHID,
                       (unsigned)(uintptr_t)((ch & 1) ? kvr0 : kvr1));
        __builtin_amdgcn_s_wait_tensorcnt(1);   // current chunk's TDM done
      } else {
        __builtin_amdgcn_s_wait_tensorcnt(0);
      }
    }
    __syncthreads();                       // kvcur visible to all waves
#else
    bf16* kvcur = kvr0;
    __syncthreads();
    for (int it = threadIdx.x; it < (32 * 512) / 8; it += 128) {
      int mm = it >> 6, dd = (it & 63) * 8;
      v8bf v = *reinterpret_cast<const v8bf*>(kvb + (size_t)(ch * 32 + mm) * DHID + dd);
      *reinterpret_cast<v8bf*>(kvcur + mm * KVR_STRIDE + dd) = v;
    }
    __syncthreads();
#endif
    // build transposed copy kvt[d][m] for the V operand (LDS -> LDS)
    for (int it = threadIdx.x; it < (32 * 512) / 8; it += 128) {
      int mm = it >> 6, dd = (it & 63) * 8;
      v8bf v = *reinterpret_cast<const v8bf*>(kvcur + mm * KVR_STRIDE + dd);
#pragma unroll
      for (int j = 0; j < 8; ++j) kvt[(dd + j) * KVT_STRIDE + mm] = v[j];
    }
    __syncthreads();

    // ---- scores: 16 rows x 32 chunk-columns, K operand from padded LDS
    v8f s0 = zero, s1 = zero;
    const bf16* kr0 = kvcur + (size_t)c * KVR_STRIDE;
    const bf16* kr1 = kvcur + (size_t)(16 + c) * KVR_STRIDE;
#pragma unroll 4
    for (int k = 0; k < DHID / 32; ++k) {
      int da = k * 32 + g * 16;
      int db = k * 32 + g * 16;
      v16bf a = load_pair(qrow + da, qrow + da + 8);
      s0 = wmma_bf16(a, load_b(kr0 + db), s0);
      s1 = wmma_bf16(a, load_b(kr1 + db), s1);
    }

    // ---- mask + online softmax (rows live across 16-lane half-waves)
    bool mA = mrow[ch * 32 + c] != 0;
    bool mB = mrow[ch * 32 + 16 + c] != 0;
    float corr[8];
#pragma unroll
    for (int i = 0; i < 8; ++i) {
      float a0 = mA ? NI : s0[i];
      float a1 = mB ? NI : s1[i];
      float mx = fmaxf(a0, a1);
#pragma unroll
      for (int off = 1; off < 16; off <<= 1) mx = fmaxf(mx, __shfl_xor(mx, off, 16));
      float nm = fmaxf(rm[i], mx);
      float e0 = __expf(a0 - nm);
      float e1 = __expf(a1 - nm);
      float sum = e0 + e1;
#pragma unroll
      for (int off = 1; off < 16; off <<= 1) sum += __shfl_xor(sum, off, 16);
      float cr = __expf(rm[i] - nm);
      rs[i] = rs[i] * cr + sum;
      rm[i] = nm;
      corr[i] = cr;
      pbuf[((size_t)wave * 16 + g * 8 + i) * KVT_STRIDE + c]      = f2bf(e0);
      pbuf[((size_t)wave * 16 + g * 8 + i) * KVT_STRIDE + 16 + c] = f2bf(e1);
    }

    // rescale running O, then accumulate P @ V-chunk
#pragma unroll
    for (int t = 0; t < 16; ++t)
#pragma unroll
      for (int i = 0; i < 8; ++i) o[t][i] *= corr[i];

    const bf16* prow = pbuf + ((size_t)wave * 16 + c) * KVT_STRIDE;
    v16bf pa = load_pair(prow + g * 16, prow + g * 16 + 8);
#pragma unroll
    for (int t = 0; t < 16; ++t) {
      int dcol = dhalf * 256 + t * 16 + c;
      const bf16* vcol = kvt + (size_t)dcol * KVT_STRIDE;
      v16bf vb = load_pair(vcol + g * 16, vcol + g * 16 + 8);
      o[t] = wmma_bf16(pa, vb, o[t]);
    }
  }

  // ---- normalize and emit fused (bf16 for the final GEMM's A operand)
#pragma unroll
  for (int t = 0; t < 16; ++t)
#pragma unroll
    for (int i = 0; i < 8; ++i) {
      float v = o[t][i] / rs[i];
      fused[(size_t)(b * L1 + m0 + g * 8 + i) * DHID + dhalf * 256 + t * 16 + c] =
          f2bf(v);
    }
}

// ---- kernel 3: out = tanh([in1 | fused] @ W2^T + b2), pipelined ----------

__global__ void __launch_bounds__(128) k_out(const bf16* __restrict__ A1,   // in1 bf16
                                             const bf16* __restrict__ A2,   // fused bf16
                                             const bf16* __restrict__ W,    // [512][1024]
                                             const float* __restrict__ b2,
                                             float* __restrict__ out) {
  int wave = threadIdx.x >> 5, lane = threadIdx.x & 31;
  int gw = blockIdx.x * 4 + wave;
  int strip = gw >> 3, nb = gw & 7;
  int m0 = strip * 16, n0 = nb * 64;
  int c = lane & 15, g = lane >> 4;

  const bf16* a1row = A1 + (size_t)(m0 + c) * DIN;
  const bf16* a2row = A2 + (size_t)(m0 + c) * DHID - DIN;  // concat on the fly
  const bf16* brow[4];
#pragma unroll
  for (int nt = 0; nt < 4; ++nt) brow[nt] = W + (size_t)(n0 + nt * 16 + c) * DCAT;

  v8f zero = {0, 0, 0, 0, 0, 0, 0, 0};
  v8f acc[4] = {zero, zero, zero, zero};

  v16bf a_cur = load_pair(a1row + g * 16, a1row + g * 16 + 8);
  v16bf b_cur[4];
#pragma unroll
  for (int nt = 0; nt < 4; ++nt) b_cur[nt] = load_b(brow[nt] + g * 16);

#pragma unroll
  for (int k = 0; k < DCAT / 32; ++k) {
    v16bf a_nxt = a_cur;
    v16bf b_nxt[4] = {b_cur[0], b_cur[1], b_cur[2], b_cur[3]};
    if (k + 1 < DCAT / 32) {
      int kn = k + 1;
      int da = kn * 32 + g * 16, db = kn * 32 + g * 16;
      const bf16* ar = (kn < DIN / 32) ? a1row : a2row;
      a_nxt = load_pair(ar + da, ar + da + 8);
#pragma unroll
      for (int nt = 0; nt < 4; ++nt) b_nxt[nt] = load_b(brow[nt] + db);
    }
#pragma unroll
    for (int nt = 0; nt < 4; ++nt) acc[nt] = wmma_bf16(a_cur, b_cur[nt], acc[nt]);
    a_cur = a_nxt;
#pragma unroll
    for (int nt = 0; nt < 4; ++nt) b_cur[nt] = b_nxt[nt];
  }
#pragma unroll
  for (int nt = 0; nt < 4; ++nt) {
    int col = n0 + nt * 16 + c;
    float bias = b2[col];
#pragma unroll
    for (int i = 0; i < 8; ++i)
      out[(size_t)(m0 + g * 8 + i) * DIN + col] = tanhf(acc[nt][i] + bias);
  }
}

// ---- launcher ------------------------------------------------------------

extern "C" void kernel_launch(void* const* d_in, const int* in_sizes, int n_in,
                              void* d_out, int out_size, void* d_ws, size_t ws_size,
                              hipStream_t stream) {
  (void)in_sizes; (void)n_in; (void)out_size; (void)ws_size;

  const float* in1 = (const float*)d_in[0];
  const float* in2 = (const float*)d_in[1];
  const unsigned char* mask2 = (const unsigned char*)d_in[2];  // jnp bool -> 1 byte
  const float* W1 = (const float*)d_in[3];
  const float* W2 = (const float*)d_in[4];
  const float* b2 = (const float*)d_in[5];
  float* out = (float*)d_out;

  const size_t n_in1 = (size_t)MTOT * DIN;       // 8,388,608
  const size_t n_in2 = (size_t)Bb * L2 * DHID;   // 8,388,608
  char* ws = (char*)d_ws;
  bf16* in1bf   = (bf16*)ws; ws += n_in1 * 2;
  bf16* in2bf   = (bf16*)ws; ws += n_in2 * 2;
  bf16* W1bf    = (bf16*)ws; ws += (size_t)DHID * DIN * 2;
  bf16* W2bf    = (bf16*)ws; ws += (size_t)DIN * DCAT * 2;
  bf16* Qbf     = (bf16*)ws; ws += n_in1 * 2;
  bf16* fusedbf = (bf16*)ws; ws += n_in1 * 2;    // total ~65.5 MiB of d_ws

  auto cvt = [&](const float* s, bf16* d, size_t n) {
    int n4 = (int)(n / 4);
    k_cvt_bf16<<<(n4 + 255) / 256, 256, 0, stream>>>(s, d, n4);
  };
  cvt(in1, in1bf, n_in1);
  cvt(in2, in2bf, n_in2);
  cvt(W1, W1bf, (size_t)DHID * DIN);
  cvt(W2, W2bf, (size_t)DIN * DCAT);

  // Q = in1 @ W1^T : 1024 strips x 8 n-blocks = 8192 waves
  k_gemm_q<<<2048, 128, 0, stream>>>(in1bf, W1bf, Qbf);

  // fused flash attention: TDM double-buffered chunks, (strip-pairs, batch)
  k_attn<<<dim3((L1 / 16) / 2, Bb), 128, SMEM_TOTAL, stream>>>(Qbf, in2bf, mask2,
                                                               fusedbf);

  // out = tanh([in1|fused] @ W2^T + b2)
  k_out<<<2048, 128, 0, stream>>>(in1bf, fusedbf, W2bf, b2, out);
}
